// UpSmpl_3066606649708
// MI455X (gfx1250) — matrix operand order repaired and checked by reading, not compiled, not run on hardware
//
#include <hip/hip_runtime.h>

typedef __attribute__((ext_vector_type(16))) __bf16 v16bf;
typedef __attribute__((ext_vector_type(8)))  float  v8f;
typedef unsigned int   u32;
typedef unsigned short u16;
typedef __attribute__((ext_vector_type(4))) u32 u32x4;
typedef __attribute__((ext_vector_type(4))) int i32x4;

#define IC     64
#define D_IN   17
#define HW     128
#define OCT    256        // OC*FCT
#define KT     1728       // IC*27
#define NKS    27         // K steps of 64 (one 3x3x3 tap each)
#define DP     19
#define HP     130
#define WP     130
#define BM     128
#define BN     128
#define ROWP   72         // 64 data + 8 pad u16 per row (144B, 16B aligned)
#define LDSBYTES (2 * (BM + BN) * ROWP * 2)   // 73728 B dynamic LDS

#if __has_builtin(__builtin_amdgcn_global_load_async_to_lds_b128) && \
    __has_builtin(__builtin_amdgcn_s_wait_asynccnt)
#define USE_ASYNC 1
typedef __attribute__((address_space(1))) i32x4* g_i32x4p;   // global (as1)
typedef __attribute__((address_space(3))) i32x4* l_i32x4p;   // LDS    (as3)

// issue NCH async 16B copies; chunk offset is a template constant so the
// builtin's IOFFSET argument (applies to BOTH global and LDS address) is an ICE
template <int C, int NCH>
__device__ __forceinline__ void async_row(const u16* g, u16* l) {
    __builtin_amdgcn_global_load_async_to_lds_b128(
        (g_i32x4p)g, (l_i32x4p)l, C * 16, 0);
    if constexpr (C + 1 < NCH) async_row<C + 1, NCH>(g, l);
}
#endif

__device__ __forceinline__ u16 f32_to_bf16(float f) {
    u32 u = __builtin_bit_cast(u32, f);
    u32 r = (u + 0x7FFFu + ((u >> 16) & 1u)) >> 16;   // round-to-nearest-even
    return (u16)r;
}

// per-K-step constant offset into x_pad (u16 units); folds under full unroll
__device__ __forceinline__ constexpr int koff(int ks) {
    int kd = ks / 9, kh = (ks % 9) / 3, kw = ks % 3;
    return ((kd * HP + kh) * WP + kw) * IC;
}

// --- Kernel 1: W (OCT, IC, 3,3,3) f32 -> Wb[o][tap*64+ic] bf16 -----------
__global__ void wconv_kernel(const float* __restrict__ W, u16* __restrict__ Wb) {
    int t = blockIdx.x * blockDim.x + threadIdx.x;
    if (t >= OCT * KT) return;
    int o   = t / KT;
    int r   = t - o * KT;
    int tap = r >> 6;
    int ic  = r & 63;
    Wb[t] = f32_to_bf16(W[(size_t)o * KT + ic * 27 + tap]);
}

// --- Kernel 2: x (IC,17,128,128) f32 -> x_pad[19][130][130][64] bf16 -----
__global__ void xpad_kernel(const float* __restrict__ x, u16* __restrict__ xp) {
    int t = blockIdx.x * blockDim.x + threadIdx.x;
    if (t >= DP * HP * WP) return;
    int wq = t % WP;
    int r  = t / WP;
    int hq = r % HP;
    int dq = r / HP;
    int d = dq - 1, h = hq - 1, w = wq - 1;
    u16* dst = xp + (size_t)t * IC;
    if (d >= 0 && d < D_IN && h >= 0 && h < HW && w >= 0 && w < HW) {
        const float* src = x + ((size_t)d * HW + h) * HW + w;
        #pragma unroll 8
        for (int ic = 0; ic < IC; ++ic)
            dst[ic] = f32_to_bf16(src[(size_t)ic * (D_IN * HW * HW)]);
    } else {
        #pragma unroll 8
        for (int ic = 0; ic < IC; ++ic) dst[ic] = 0;
    }
}

// --- Kernel 3: implicit GEMM conv + bias + pixel-shuffle + residual ------
union FragU { u32x4 u[2]; v16bf v; };

__global__ __launch_bounds__(128)
void conv_gemm_kernel(const u16* __restrict__ Wb, const u16* __restrict__ xp,
                      const float* __restrict__ x, const float* __restrict__ bias,
                      float* __restrict__ out) {
    extern __shared__ __align__(16) u16 smem[];   // 73728 B dynamic
    u16* Abase[2] = { smem,                 smem + BM * ROWP };
    u16* Bbase[2] = { smem + 2 * BM * ROWP, smem + 2 * BM * ROWP + BN * ROWP };

    const int t      = threadIdx.x;     // 128 threads = 4 waves
    const int lane   = t & 31;
    const int wave   = t >> 5;
    const int lane16 = lane & 15;
    const int hi     = lane >> 4;
    const int wm     = wave >> 1;   // 0..1 : 64-row M strip
    const int wn     = wave & 1;    // 0..1 : 64-col N strip

    // block -> output coordinates (one block = full w row of one (d,hh))
    const int nt    = blockIdx.x;       // 0..2175
    const int d     = nt >> 7;          // 0..16
    const int hh    = nt & 127;         // 0..127
    const int mBase = blockIdx.y << 7;  // 0 or 128

    // staging: thread t owns A row t and B row t (8 x 16B chunks each)
    const u16* aSrc = Wb + (size_t)(mBase + t) * KT;
    const u16* bSrc = xp + (((size_t)d * HP + hh) * WP + t) * IC;
    u16* aDst[2] = { Abase[0] + t * ROWP, Abase[1] + t * ROWP };
    u16* bDst[2] = { Bbase[0] + t * ROWP, Bbase[1] + t * ROWP };

#ifdef USE_ASYNC
    auto phase1 = [&](int ks, int buf) {  // issue async global->LDS copies
        async_row<0, 8>(aSrc + ks * 64, aDst[buf]);
        async_row<0, 8>(bSrc + koff(ks), bDst[buf]);
    };
    auto phase2 = [&](int buf) { (void)buf; };
    auto fence  = [&]() {
        __builtin_amdgcn_s_wait_asynccnt(0);
        __syncthreads();
    };
#else
    u32x4 regA[8], regB[8];
    auto phase1 = [&](int ks, int buf) {  // global -> regs
        (void)buf;
        const u16* ga = aSrc + ks * 64;
        const u16* gb = bSrc + koff(ks);
        #pragma unroll
        for (int c = 0; c < 8; ++c) regA[c] = *(const u32x4*)(ga + c * 8);
        #pragma unroll
        for (int c = 0; c < 8; ++c) regB[c] = *(const u32x4*)(gb + c * 8);
    };
    auto phase2 = [&](int buf) {          // regs -> LDS
        #pragma unroll
        for (int c = 0; c < 8; ++c) *(u32x4*)(aDst[buf] + c * 8) = regA[c];
        #pragma unroll
        for (int c = 0; c < 8; ++c) *(u32x4*)(bDst[buf] + c * 8) = regB[c];
    };
    auto fence = [&]() { __syncthreads(); };
#endif

    v8f acc[4][4] = {};   // 64x64 per wave: 4 M-frags x 4 N-frags

    auto compute = [&](int buf) {
        #pragma unroll
        for (int kk = 0; kk < 64; kk += 32) {
            FragU a[4], b[4];
            // A 16x32 bf16: lanes<16: K0-7 @+0, K16-23 @+32B; lanes>=16: +16B
            #pragma unroll
            for (int mi = 0; mi < 4; ++mi) {
                const u16* Ap = Abase[buf] + (wm * 64 + mi * 16 + lane16) * ROWP + kk;
                a[mi].u[0] = *(const u32x4*)(Ap + hi * 8);
                a[mi].u[1] = *(const u32x4*)(Ap + 16 + hi * 8);
            }
            // B 32x16 bf16 (N-major rows): lanes<16: K0-7 @+0, K16-23 @+32B; lanes>=16: +16B
            #pragma unroll
            for (int ni = 0; ni < 4; ++ni) {
                const u16* Bp = Bbase[buf] + (wn * 64 + ni * 16 + lane16) * ROWP + kk;
                b[ni].u[0] = *(const u32x4*)(Bp + hi * 8);
                b[ni].u[1] = *(const u32x4*)(Bp + 16 + hi * 8);
            }
            #pragma unroll
            for (int mi = 0; mi < 4; ++mi)
            #pragma unroll
            for (int ni = 0; ni < 4; ++ni)
                acc[mi][ni] = __builtin_amdgcn_wmma_f32_16x16x32_bf16(
                    false, a[mi].v, false, b[ni].v, (short)0, acc[mi][ni], false, false);
        }
    };

    // ---- fully unrolled, double-buffered K loop (all offsets constant) ----
    phase1(0, 0);
    phase2(0);
    fence();
    #pragma unroll
    for (int ks = 0; ks < NKS; ++ks) {
        if (ks + 1 < NKS) phase1(ks + 1, (ks + 1) & 1);
        if (ks + 2 < NKS)   // keep a global_prefetch_b8 ahead of the B stream
            __builtin_prefetch(bSrc + koff(ks + 2), 0, 1);
        compute(ks & 1);
        if (ks + 1 < NKS) {
            phase2((ks + 1) & 1);
            fence();
        }
    }

    // ---- fused epilogue: bias + pixel shuffle + residual ----
    #pragma unroll
    for (int mi = 0; mi < 4; ++mi)
    #pragma unroll
    for (int ni = 0; ni < 4; ++ni)
    #pragma unroll
    for (int v = 0; v < 8; ++v) {
        int o  = mBase + wm * 64 + mi * 16 + v + hi * 8;   // conv output channel
        int ww = wn * 64 + ni * 16 + lane16;               // conv output w
        float val = acc[mi][ni][v] + bias[o];
        int nc, frame, s1, s2, ric, rd;
        if (d == 0) {
            nc = o & 63;
            if (nc >= 32) continue;              // hf keeps lower half only
            s2 = (o >> 6) & 1;
            s1 = (o >> 7) & 1;
            frame = 0;
            ric = (s1 * 64 + s2 * 32 + nc) >> 1; // repeat(x, 2) inverse
            rd  = 0;
        } else {
            nc = o & 31;
            s2 = (o >> 5) & 1;
            s1 = (o >> 6) & 1;
            int r1 = (o >> 7) & 1;
            frame = (d - 1) * 2 + r1 + 1;
            ric = o >> 2;                         // repeat(x, 4) inverse
            rd  = d;
        }
        float res = x[(((size_t)ric * D_IN + rd) * HW + hh) * HW + ww];
        size_t oidx = (((size_t)nc * 33 + frame) * 256 + (2 * hh + s1)) * 256
                    + (2 * ww + s2);
        out[oidx] = val + res;
    }
}

extern "C" void kernel_launch(void* const* d_in, const int* in_sizes, int n_in,
                              void* d_out, int out_size, void* d_ws, size_t ws_size,
                              hipStream_t stream) {
    const float* x = (const float*)d_in[0];
    const float* W = (const float*)d_in[1];
    const float* b = (const float*)d_in[2];
    float* out = (float*)d_out;

    u16* Wb = (u16*)d_ws;                             // 256*1728*2  ≈ 0.88 MB
    u16* xp = (u16*)((char*)d_ws + (1u << 20));       // 19*130*130*64*2 ≈ 41 MB

    // allow >64KB dynamic LDS (gfx1250 WGP has 320KB); harmless if redundant
    (void)hipFuncSetAttribute((const void*)conv_gemm_kernel,
                              hipFuncAttributeMaxDynamicSharedMemorySize, LDSBYTES);

    int nW = OCT * KT;
    wconv_kernel<<<(nW + 255) / 256, 256, 0, stream>>>(W, Wb);
    int nP = DP * HP * WP;
    xpad_kernel<<<(nP + 255) / 256, 256, 0, stream>>>(x, xp);

    dim3 grid(2176, 2);   // (d,h) rows x M tiles (256/128)
    conv_gemm_kernel<<<grid, 128, LDSBYTES, stream>>>(Wb, xp, x, b, out);
}
